// net_90168543412242
// MI455X (gfx1250) — matrix operand order repaired and checked by reading, not compiled, not run on hardware
//
#include <hip/hip_runtime.h>

typedef __attribute__((ext_vector_type(16))) _Float16 v16h;
typedef __attribute__((ext_vector_type(8)))  float    v8f;

#define B_N      128
#define SEQ_LENN 360
#define C_INN    7
#define PERIODD  30
#define SEQ_RR   12
#define HID      64
#define OUT_CH   320
#define DIMM     420
#define DEPTHH   10
#define LC_T     7
#define LC_S     14
#define NROWS_T  (B_N * LC_T)            // 896
#define NROWS_S  (B_N * SEQ_RR * LC_S)   // 21504
#define SPLITK   8

// K-tile counts (K rounded up to multiples of 32)
#define KT_T   12      // proj_t  K=360
#define KT_S   1       // proj_s  K=30
#define KT_Y1  70      // y1      K=2240
#define KT_Y2  1680    // y2      K=53760

__device__ __forceinline__ float gelu_tanh(float x) {
    const float k0 = 0.7978845608028654f; // sqrt(2/pi)
    float x3 = x * x * x;
    return 0.5f * x * (1.0f + tanhf(k0 * (x + 0.044715f * x3)));
}

// Write one fp32 value into the split-f16 packed A-operand layout
// (contiguous k = 16*hiK + e, same as packB): lane = (hiK<<4)|(row&15),
// K-in-tile w -> element e = w&15, hiK = (w>>4)&1.
__device__ __forceinline__ void packA_write(_Float16* __restrict__ hi,
                                            _Float16* __restrict__ lo,
                                            int kT, int arow, int ak, float v) {
    int mt = arow >> 4;
    int kt = ak >> 5;
    int w  = ak & 31;
    int hl = (w >> 4) & 1;
    int e  = w & 15;
    int lane = (hl << 4) | (arow & 15);
    size_t off = ((((size_t)mt * kT + kt) * 32 + lane) << 4) + e;
    _Float16 h = (_Float16)v;
    hi[off] = h;
    lo[off] = (_Float16)(v - (float)h);
}

__global__ void zero_h_kernel(_Float16* __restrict__ p, int n) {
    int i = blockIdx.x * blockDim.x + threadIdx.x;
    if (i < n) p[i] = (_Float16)0.0f;
}

// ---------------------------------------------------------------------------
// series_decomp_multi -> trend (B,360,7) fp32 (for DTW) + packed A for proj_t
// A row = b*7+c, A col(K) = l
// ---------------------------------------------------------------------------
__global__ void decomp_kernel(const float* __restrict__ x,
                              const float* __restrict__ dw,
                              const float* __restrict__ db,
                              float* __restrict__ trend,
                              _Float16* __restrict__ pa_hi,
                              _Float16* __restrict__ pa_lo) {
    int idx = blockIdx.x * blockDim.x + threadIdx.x;
    const int total = B_N * SEQ_LENN * C_INN;
    if (idx >= total) return;
    int c = idx % C_INN;
    int l = (idx / C_INN) % SEQ_LENN;
    int b = idx / (C_INN * SEQ_LENN);
    const float* xb = x + (size_t)b * SEQ_LENN * C_INN + c;
    float xv = xb[(size_t)l * C_INN];
    const int ks[3] = {4, 8, 12};
    float ma[3];
#pragma unroll
    for (int m = 0; m < 3; m++) {
        int k = ks[m];
        int front = k - 1 - (k - 1) / 2;
        float s = 0.f;
        for (int j = 0; j < k; j++) {
            int q = l + j - front;
            q = q < 0 ? 0 : (q >= SEQ_LENN ? SEQ_LENN - 1 : q);
            s += xb[(size_t)q * C_INN];
        }
        ma[m] = s / (float)k;
    }
    float lg0 = xv * dw[0] + db[0];
    float lg1 = xv * dw[1] + db[1];
    float lg2 = xv * dw[2] + db[2];
    float mx = fmaxf(lg0, fmaxf(lg1, lg2));
    float e0 = expf(lg0 - mx), e1 = expf(lg1 - mx), e2 = expf(lg2 - mx);
    float inv = 1.f / (e0 + e1 + e2);
    float t = (ma[0] * e0 + ma[1] * e1 + ma[2] * e2) * inv;
    trend[idx] = t;
    packA_write(pa_hi, pa_lo, KT_T, b * C_INN + c, l, t);
}

// ---------------------------------------------------------------------------
// xs segments -> packed A for proj_s; A row = n*14+ch, A col(K) = t
// ---------------------------------------------------------------------------
__global__ void build_xs_kernel(const float* __restrict__ x,
                                const float* __restrict__ xm,
                                _Float16* __restrict__ pa_hi,
                                _Float16* __restrict__ pa_lo) {
    int idx = blockIdx.x * blockDim.x + threadIdx.x;
    const int total = B_N * SEQ_RR * LC_S * PERIODD;
    if (idx >= total) return;
    int t = idx % PERIODD;
    int ch = (idx / PERIODD) % LC_S;
    int n = idx / (PERIODD * LC_S);
    int b = n / SEQ_RR, s = n % SEQ_RR;
    int lseq = s * PERIODD + t;
    float v;
    if (ch < C_INN) v = x[((size_t)b * SEQ_LENN + lseq) * C_INN + ch];
    else            v = xm[((size_t)b * SEQ_LENN + lseq) * C_INN + (ch - C_INN)];
    packA_write(pa_hi, pa_lo, KT_S, n * LC_S + ch, t, v);
}

// ---------------------------------------------------------------------------
// Pack weight matrix B (K,N) into split-f16 WMMA B-operand layout:
// lane = (hiK<<4)|(col&15), element e -> K = kt*32 + e + 16*(lane>=16).
// Iterates packed index space, so edges are zero-filled here.
// ---------------------------------------------------------------------------
__global__ void packB_kernel(const float* __restrict__ B, int ldb, int K, int N,
                             _Float16* __restrict__ hi, _Float16* __restrict__ lo,
                             int kT, int nT) {
    int idx = blockIdx.x * blockDim.x + threadIdx.x;
    int total = nT * kT * 512;
    if (idx >= total) return;
    int e    = idx & 15;
    int lane = (idx >> 4) & 31;
    int tile = idx >> 9;
    int nt = tile / kT, kt = tile % kT;
    int col = (nt << 4) + (lane & 15);
    int bk  = (kt << 5) + e + ((lane >> 4) << 4);
    float v = (bk < K && col < N) ? B[(size_t)bk * ldb + col] : 0.0f;
    _Float16 h = (_Float16)v;
    hi[idx] = h;
    lo[idx] = (_Float16)(v - (float)h);
}

// ---------------------------------------------------------------------------
// Fully packed WMMA GEMM: inner loop = 4x v16h load + prefetch + 3x v_wmma.
// One wave per 16x16 C tile, 4 waves/block.
// ---------------------------------------------------------------------------
__global__ void wmma_gemm_packed(const _Float16* __restrict__ PAhi,
                                 const _Float16* __restrict__ PAlo,
                                 const _Float16* __restrict__ PBhi,
                                 const _Float16* __restrict__ PBlo,
                                 float* __restrict__ C, int ldc,
                                 const float* __restrict__ bias,
                                 int M, int N, int kT) {
    int lane = threadIdx.x & 31;
    int wave = threadIdx.x >> 5;
    int nTiles = (N + 15) >> 4;
    int mTiles = (M + 15) >> 4;
    int tile = blockIdx.x * (blockDim.x >> 5) + wave;
    if (tile >= mTiles * nTiles) return;   // wave-uniform
    int mt = tile / nTiles, nt = tile % nTiles;

    const _Float16* pa_hi = PAhi + ((((size_t)mt * kT) * 32 + lane) << 4);
    const _Float16* pa_lo = PAlo + ((((size_t)mt * kT) * 32 + lane) << 4);
    const _Float16* pb_hi = PBhi + ((((size_t)nt * kT) * 32 + lane) << 4);
    const _Float16* pb_lo = PBlo + ((((size_t)nt * kT) * 32 + lane) << 4);

    v8f acc = {};
    for (int kt = 0; kt < kT; kt++) {
        v16h ah = *(const v16h*)pa_hi;
        v16h al = *(const v16h*)pa_lo;
        v16h bh = *(const v16h*)pb_hi;
        v16h bl = *(const v16h*)pb_lo;
        __builtin_prefetch(pa_hi + 512, 0, 0);
        __builtin_prefetch(pb_hi + 512, 0, 0);
        acc = __builtin_amdgcn_wmma_f32_16x16x32_f16(false, ah, false, bh,
                                                     (short)0, acc, false, false);
        acc = __builtin_amdgcn_wmma_f32_16x16x32_f16(false, al, false, bh,
                                                     (short)0, acc, false, false);
        acc = __builtin_amdgcn_wmma_f32_16x16x32_f16(false, ah, false, bl,
                                                     (short)0, acc, false, false);
        pa_hi += 512; pa_lo += 512; pb_hi += 512; pb_lo += 512;
    }

    int col = (nt << 4) + (lane & 15);
    int hiLane = lane >> 4;
    if (col < N) {
        float bv = bias ? bias[col] : 0.0f;
#pragma unroll
        for (int r = 0; r < 8; r++) {
            int row = (mt << 4) + r + (hiLane << 3);
            if (row < M) C[(size_t)row * ldc + col] = acc[r] + bv;
        }
    }
}

// ---------------------------------------------------------------------------
// Packed-A x streamed-fp32-B GEMM (for y2: Wr is 90MB, streamed once -> no
// point prepacking). Column edge handled via clamp + select, no branches.
// Split-K partials (gridDim.y), deterministic reduce afterwards.
// ---------------------------------------------------------------------------
__global__ void wmma_gemm_pA(const _Float16* __restrict__ PAhi,
                             const _Float16* __restrict__ PAlo,
                             const float* __restrict__ Bm, int ldb,
                             float* __restrict__ Cpart,
                             int M, int N, int kT, int ktChunk) {
    int lane = threadIdx.x & 31;
    int wave = threadIdx.x >> 5;
    int nTiles = (N + 15) >> 4;
    int mTiles = (M + 15) >> 4;
    int tile = blockIdx.x * (blockDim.x >> 5) + wave;
    if (tile >= mTiles * nTiles) return;
    int mt = tile / nTiles, nt = tile % nTiles;

    int ktBeg = blockIdx.y * ktChunk;
    int ktEnd = ktBeg + ktChunk;
    if (ktEnd > kT) ktEnd = kT;

    const _Float16* pa_hi = PAhi + ((((size_t)mt * kT + ktBeg) * 32 + lane) << 4);
    const _Float16* pa_lo = PAlo + ((((size_t)mt * kT + ktBeg) * 32 + lane) << 4);

    int col = (nt << 4) + (lane & 15);
    bool colOK = col < N;
    int colc = colOK ? col : N - 1;   // clamp: loads always in-bounds
    int hiLane = lane >> 4;

    v8f acc = {};
    for (int kt = ktBeg; kt < ktEnd; kt++) {
        v16h ah = *(const v16h*)pa_hi;
        v16h al = *(const v16h*)pa_lo;
        v16h bh = {}, bl = {};
        const float* bp = Bm + ((size_t)((kt << 5) + (hiLane << 4))) * ldb + colc;
#pragma unroll
        for (int e = 0; e < 16; e++) {
            float bv = bp[(size_t)e * ldb];
            bv = colOK ? bv : 0.0f;             // v_cndmask, no exec juggling
            _Float16 h = (_Float16)bv;
            bh[e] = h;
            bl[e] = (_Float16)(bv - (float)h);
        }
        __builtin_prefetch(pa_hi + 512, 0, 0);
        acc = __builtin_amdgcn_wmma_f32_16x16x32_f16(false, ah, false, bh,
                                                     (short)0, acc, false, false);
        acc = __builtin_amdgcn_wmma_f32_16x16x32_f16(false, al, false, bh,
                                                     (short)0, acc, false, false);
        acc = __builtin_amdgcn_wmma_f32_16x16x32_f16(false, ah, false, bl,
                                                     (short)0, acc, false, false);
        pa_hi += 512; pa_lo += 512;
    }

    float* Cout = Cpart + (size_t)blockIdx.y * M * N;
    if (colOK) {
#pragma unroll
        for (int r = 0; r < 8; r++) {
            int row = (mt << 4) + r + (hiLane << 3);
            if (row < M) Cout[(size_t)row * N + col] = acc[r];
        }
    }
}

// ---------------------------------------------------------------------------
// Residual dilated conv layer (64->64, k=3, SAME, gelu residual)
// ---------------------------------------------------------------------------
__global__ void conv_layer_kernel(const float* __restrict__ Hin,
                                  float* __restrict__ Hout,
                                  const float* __restrict__ w,   // (64,64,3)
                                  const float* __restrict__ bias,
                                  int nRows, int Lc, int dil) {
    int idx = blockIdx.x * blockDim.x + threadIdx.x;
    if (idx >= nRows * HID) return;
    int o = idx & (HID - 1);
    int row = idx >> 6;
    int p = row % Lc;
    int base = row - p;
    float acc = bias[o];
#pragma unroll
    for (int t = 0; t < 3; t++) {
        int q = p + (t - 1) * dil;
        if (q >= 0 && q < Lc) {
            const float* hq = Hin + (size_t)(base + q) * HID;
            const float* wt = w + (size_t)o * HID * 3 + t;
            for (int i = 0; i < HID; i++)
                acc = fmaf(wt[i * 3], hq[i], acc);
        }
    }
    Hout[idx] = Hin[idx] + gelu_tanh(acc);
}

// Final conv (64->320, k=3, dil 1) writing DIRECTLY into packed-A layout of
// the following GEMM: arow = row / rowsPerBatch, ak = (row%rowsPerBatch)*320+o
__global__ void final_conv_pack_kernel(const float* __restrict__ Hin,
                                       const float* __restrict__ w,   // (320,64,3)
                                       const float* __restrict__ bias,
                                       int nRows, int Lc, int rowsPerBatch, int kT,
                                       _Float16* __restrict__ pa_hi,
                                       _Float16* __restrict__ pa_lo) {
    int idx = blockIdx.x * blockDim.x + threadIdx.x;
    if (idx >= nRows * OUT_CH) return;
    int o = idx % OUT_CH;
    int row = idx / OUT_CH;
    int p = row % Lc;
    int base = row - p;
    float acc = bias[o];
#pragma unroll
    for (int t = 0; t < 3; t++) {
        int q = p + (t - 1);
        if (q >= 0 && q < Lc) {
            const float* hq = Hin + (size_t)(base + q) * HID;
            const float* wt = w + (size_t)o * HID * 3 + t;
            for (int i = 0; i < HID; i++)
                acc = fmaf(wt[i * 3], hq[i], acc);
        }
    }
    int arow = row / rowsPerBatch;
    int ak = (row % rowsPerBatch) * OUT_CH + o;
    packA_write(pa_hi, pa_lo, kT, arow, ak, acc);
}

// ---------------------------------------------------------------------------
// Exact DTW DP per (segment-batch, channel)
// ---------------------------------------------------------------------------
__global__ void dtw_kernel(const float* __restrict__ x,
                           const float* __restrict__ trend,
                           float* __restrict__ ndtw) {
    int r = blockIdx.x * blockDim.x + threadIdx.x;
    if (r >= B_N * C_INN) return;
    int i = r / C_INN, c = r % C_INN;
    int b = i / SEQ_RR, s = i % SEQ_RR;
    float a[PERIODD];
    for (int t = 0; t < PERIODD; t++)
        a[t] = x[((size_t)b * SEQ_LENN + s * PERIODD + t) * C_INN + c];
    const float* bb = trend + (size_t)i * SEQ_LENN * C_INN + c;
    float prev[SEQ_LENN];
    float run = 0.f;
    for (int j = 0; j < SEQ_LENN; j++) {
        run += fabsf(a[0] - bb[(size_t)j * C_INN]);
        prev[j] = run;
    }
    const float BIG = 1e30f;
    for (int ii = 1; ii < PERIODD; ii++) {
        float ai = a[ii];
        float diag = BIG, left = BIG;
        for (int j = 0; j < SEQ_LENN; j++) {
            float up = prev[j];
            float cst = fabsf(ai - bb[(size_t)j * C_INN]);
            float d = cst + fminf(fminf(up, diag), left);
            prev[j] = d;
            diag = up;
            left = d;
        }
    }
    ndtw[r] = 1.0f / (1.0f + expf(-(float)SEQ_LENN / prev[SEQ_LENN - 1]));
}

__global__ void reduce_y2_kernel(const float* __restrict__ partial,
                                 const float* __restrict__ bias,
                                 float* __restrict__ y2) {
    int idx = blockIdx.x * blockDim.x + threadIdx.x;
    const int total = B_N * DIMM;
    if (idx >= total) return;
    float s = bias[idx % DIMM];
    for (int k = 0; k < SPLITK; k++)
        s += partial[(size_t)k * total + idx];
    y2[idx] = s;
}

__global__ void combine_kernel(const float* __restrict__ y1,
                               const float* __restrict__ y2,
                               const float* __restrict__ g1,
                               const float* __restrict__ g2,
                               float* __restrict__ out0) {
    int idx = blockIdx.x * blockDim.x + threadIdx.x;
    if (idx >= B_N * DIMM) return;
    out0[idx] = y1[idx] * g1[0] + y2[idx] * g2[0];
}

// ---------------------------------------------------------------------------
extern "C" void kernel_launch(void* const* d_in, const int* in_sizes, int n_in,
                              void* d_out, int out_size, void* d_ws, size_t ws_size,
                              hipStream_t stream) {
    (void)in_sizes; (void)n_in; (void)out_size; (void)ws_size;

    const float* x        = (const float*)d_in[0];
    const float* x_mark   = (const float*)d_in[1];
    const float* g1       = (const float*)d_in[2];
    const float* g2       = (const float*)d_in[3];
    const float* decomp_w = (const float*)d_in[4];
    const float* decomp_b = (const float*)d_in[5];
    const float* Wp_t     = (const float*)d_in[6];
    const float* bp_t     = (const float*)d_in[7];
    const float* conv_w_t = (const float*)d_in[8];
    const float* conv_b_t = (const float*)d_in[9];
    const float* fin_w_t  = (const float*)d_in[10];
    const float* fin_b_t  = (const float*)d_in[11];
    const float* Wp_s     = (const float*)d_in[12];
    const float* bp_s     = (const float*)d_in[13];
    const float* conv_w_s = (const float*)d_in[14];
    const float* conv_b_s = (const float*)d_in[15];
    const float* fin_w_s  = (const float*)d_in[16];
    const float* fin_b_s  = (const float*)d_in[17];
    const float* Wr_time  = (const float*)d_in[18];
    const float* br_time  = (const float*)d_in[19];
    const float* Wr       = (const float*)d_in[20];
    const float* br       = (const float*)d_in[21];

    float* out  = (float*)d_out;
    float* out0 = out;
    float* y1   = out + 53760;
    float* y2   = out + 107520;
    float* ndtw = out + 161280;

    // --- workspace carve-out (1KB aligned) ---
    char* base = (char*)d_ws;
    size_t off = 0;
    auto alloc = [&](size_t bytes) -> void* {
        off = (off + 1023) & ~(size_t)1023;
        void* p = base + off;
        off += bytes;
        return p;
    };
    const int SZ_PAT = (NROWS_T / 16) * KT_T * 512;    // 344064 halves
    const int SZ_PAS = (NROWS_S / 16) * KT_S * 512;    // 688128
    const int SZ_PBT = 4 * KT_T * 512;                 // 24576
    const int SZ_PBS = 4 * KT_S * 512;                 // 2048
    const int SZ_PA1 = (B_N / 16) * KT_Y1 * 512;       // 286720
    const int SZ_PB1 = 27 * KT_Y1 * 512;               // 967680
    const int SZ_PA2 = (B_N / 16) * KT_Y2 * 512;       // 6881280

    float*    trend  = (float*)alloc((size_t)B_N * SEQ_LENN * C_INN * 4);
    float*    Ht0    = (float*)alloc((size_t)NROWS_T * HID * 4);
    float*    Ht1    = (float*)alloc((size_t)NROWS_T * HID * 4);
    float*    Hs0    = (float*)alloc((size_t)NROWS_S * HID * 4);
    float*    Hs1    = (float*)alloc((size_t)NROWS_S * HID * 4);
    float*    part   = (float*)alloc((size_t)SPLITK * B_N * DIMM * 4);
    _Float16* PAt_hi = (_Float16*)alloc((size_t)SZ_PAT * 2);
    _Float16* PAt_lo = (_Float16*)alloc((size_t)SZ_PAT * 2);
    _Float16* PAs_hi = (_Float16*)alloc((size_t)SZ_PAS * 2);
    _Float16* PAs_lo = (_Float16*)alloc((size_t)SZ_PAS * 2);
    _Float16* PBt_hi = (_Float16*)alloc((size_t)SZ_PBT * 2);
    _Float16* PBt_lo = (_Float16*)alloc((size_t)SZ_PBT * 2);
    _Float16* PBs_hi = (_Float16*)alloc((size_t)SZ_PBS * 2);
    _Float16* PBs_lo = (_Float16*)alloc((size_t)SZ_PBS * 2);
    _Float16* PA1_hi = (_Float16*)alloc((size_t)SZ_PA1 * 2);
    _Float16* PA1_lo = (_Float16*)alloc((size_t)SZ_PA1 * 2);
    _Float16* PB1_hi = (_Float16*)alloc((size_t)SZ_PB1 * 2);
    _Float16* PB1_lo = (_Float16*)alloc((size_t)SZ_PB1 * 2);
    _Float16* PA2_hi = (_Float16*)alloc((size_t)SZ_PA2 * 2);
    _Float16* PA2_lo = (_Float16*)alloc((size_t)SZ_PA2 * 2);

    // 0) zero-fill packed A buffers whose K isn't a multiple of 32
    zero_h_kernel<<<(SZ_PAT + 255) / 256, 256, 0, stream>>>(PAt_hi, SZ_PAT);
    zero_h_kernel<<<(SZ_PAT + 255) / 256, 256, 0, stream>>>(PAt_lo, SZ_PAT);
    zero_h_kernel<<<(SZ_PAS + 255) / 256, 256, 0, stream>>>(PAs_hi, SZ_PAS);
    zero_h_kernel<<<(SZ_PAS + 255) / 256, 256, 0, stream>>>(PAs_lo, SZ_PAS);

    // 1) decomposition -> trend + packed A(proj_t)
    {
        int total = B_N * SEQ_LENN * C_INN;
        decomp_kernel<<<(total + 255) / 256, 256, 0, stream>>>(
            x, decomp_w, decomp_b, trend, PAt_hi, PAt_lo);
    }
    // 2) segment inputs -> packed A(proj_s)
    {
        int total = B_N * SEQ_RR * LC_S * PERIODD;
        build_xs_kernel<<<(total + 255) / 256, 256, 0, stream>>>(x, x_mark, PAs_hi, PAs_lo);
    }
    // 3) pack the reused weight matrices
    packB_kernel<<<(4 * KT_T * 512 + 255) / 256, 256, 0, stream>>>(
        Wp_t, HID, SEQ_LENN, HID, PBt_hi, PBt_lo, KT_T, 4);
    packB_kernel<<<(4 * KT_S * 512 + 255) / 256, 256, 0, stream>>>(
        Wp_s, HID, PERIODD, HID, PBs_hi, PBs_lo, KT_S, 4);
    packB_kernel<<<(27 * KT_Y1 * 512 + 255) / 256, 256, 0, stream>>>(
        Wr_time, DIMM, 2240, DIMM, PB1_hi, PB1_lo, KT_Y1, 27);

    // 4) projections (fully packed WMMA GEMMs)
    {
        int tiles = (NROWS_T / 16) * (HID / 16);      // 224
        wmma_gemm_packed<<<dim3((tiles + 3) / 4), 128, 0, stream>>>(
            PAt_hi, PAt_lo, PBt_hi, PBt_lo, Ht0, HID, bp_t, NROWS_T, HID, KT_T);
    }
    {
        int tiles = (NROWS_S / 16) * (HID / 16);      // 5376
        wmma_gemm_packed<<<dim3((tiles + 3) / 4), 128, 0, stream>>>(
            PAs_hi, PAs_lo, PBs_hi, PBs_lo, Hs0, HID, bp_s, NROWS_S, HID, KT_S);
    }
    // 5) residual dilated conv stacks, final conv emits packed A directly
    {
        float* a = Ht0; float* b = Ht1;
        int total = NROWS_T * HID;
        for (int i = 0; i < DEPTHH; i++) {
            conv_layer_kernel<<<(total + 255) / 256, 256, 0, stream>>>(
                a, b, conv_w_t + (size_t)i * HID * HID * 3, conv_b_t + (size_t)i * HID,
                NROWS_T, LC_T, 1 << i);
            float* tpt = a; a = b; b = tpt;
        }
        final_conv_pack_kernel<<<(NROWS_T * OUT_CH + 255) / 256, 256, 0, stream>>>(
            a, fin_w_t, fin_b_t, NROWS_T, LC_T, LC_T, KT_Y1, PA1_hi, PA1_lo);
    }
    {
        float* a = Hs0; float* b = Hs1;
        int total = NROWS_S * HID;
        for (int i = 0; i < DEPTHH; i++) {
            conv_layer_kernel<<<(total + 255) / 256, 256, 0, stream>>>(
                a, b, conv_w_s + (size_t)i * HID * HID * 3, conv_b_s + (size_t)i * HID,
                NROWS_S, LC_S, 1 << i);
            float* tpt = a; a = b; b = tpt;
        }
        final_conv_pack_kernel<<<(NROWS_S * OUT_CH + 255) / 256, 256, 0, stream>>>(
            a, fin_w_s, fin_b_s, NROWS_S, LC_S, SEQ_RR * LC_S, KT_Y2, PA2_hi, PA2_lo);
    }
    // 6) y1 = rep_t(128x2240) @ Wr_time + br_time   (fully packed)
    {
        int tiles = (B_N / 16) * ((DIMM + 15) / 16);  // 216
        wmma_gemm_packed<<<dim3((tiles + 3) / 4), 128, 0, stream>>>(
            PA1_hi, PA1_lo, PB1_hi, PB1_lo, y1, DIMM, br_time, B_N, DIMM, KT_Y1);
    }
    // 7) y2 = rep2(128x53760) @ Wr + br  (packed A, streamed fp32 B, split-K)
    {
        int tiles = (B_N / 16) * ((DIMM + 15) / 16);  // 216
        wmma_gemm_pA<<<dim3((tiles + 3) / 4, SPLITK), 128, 0, stream>>>(
            PA2_hi, PA2_lo, Wr, DIMM, part, B_N, DIMM, KT_Y2, KT_Y2 / SPLITK);
        reduce_y2_kernel<<<(B_N * DIMM + 255) / 256, 256, 0, stream>>>(part, br, y2);
    }
    // 8) DTW head
    dtw_kernel<<<(B_N * C_INN + 127) / 128, 128, 0, stream>>>(x, trend, ndtw);
    // 9) out0 = sg(y1)*g1 + sg(y2)*g2
    combine_kernel<<<(B_N * DIMM + 255) / 256, 256, 0, stream>>>(y1, y2, g1, g2, out0);
}
